// NeuralFingerprint_56710748176715
// MI455X (gfx1250) — matrix operand order repaired and checked by reading, not compiled, not run on hardware
//
#include <hip/hip_runtime.h>
#include <hip/hip_bf16.h>

// ---------------------------------------------------------------------------
// NeuralFingerprint on gfx1250 (MI455X)
//   - bf16-in / f32-accumulate WMMA for all GEMMs
//   - bf16 LDS staging (convert once, A-fragments = pure ds_load_b128)
//   - h1/h2 stored bf16 (halves L2 gather traffic)
//   - async global->LDS staging where rows are raw bf16 copies
//   - run-length-reduced segment sum into fingerprint
// ---------------------------------------------------------------------------

typedef unsigned short bf16_t;
typedef __attribute__((ext_vector_type(16))) __bf16          v16bf;
typedef __attribute__((ext_vector_type(8)))  float           v8f;
typedef __attribute__((ext_vector_type(8)))  unsigned short  v8us;
typedef __attribute__((ext_vector_type(4)))  int             v4i;
typedef __attribute__((address_space(1))) v4i* as1_v4i_ptr;  // global <4 x i32>*
typedef __attribute__((address_space(3))) v4i* as3_v4i_ptr;  // LDS    <4 x i32>*

#define N_ATOMS 200000
#define B_MOL   8192
#define EDGE_F  16
#define ND_B    50000

#if __has_builtin(__builtin_amdgcn_global_load_async_to_lds_b128)
#define USE_ASYNC_LDS 1
#else
#define USE_ASYNC_LDS 0
#endif

__device__ __forceinline__ void wait_asynccnt0() {
#if __has_builtin(__builtin_amdgcn_s_wait_asynccnt)
  __builtin_amdgcn_s_wait_asynccnt(0);
#else
  asm volatile("s_wait_asynccnt 0x0" ::: "memory");
#endif
}

__device__ __forceinline__ bf16_t f2bf(float f) {
  __bf16 h = (__bf16)f;                       // RNE convert (v_cvt path)
  return __builtin_bit_cast(bf16_t, h);
}
__device__ __forceinline__ float bf2f(bf16_t b) {
  unsigned int u = ((unsigned int)b) << 16;
  return __builtin_bit_cast(float, u);
}
__device__ __forceinline__ float xval(const float* p, size_t i) { return p[i]; }
__device__ __forceinline__ float xval(const bf16_t* p, size_t i) { return bf2f(p[i]); }

// contiguous row staging into bf16 LDS ------------------------------------
__device__ __forceinline__ void stage_block(bf16_t* dst, const float* src,
                                            int nelem, int t) {
  for (int i = t; i < nelem; i += 256) dst[i] = f2bf(src[i]);
}
__device__ __forceinline__ void stage_block(bf16_t* dst, const bf16_t* src,
                                            int nelem, int t) {
  const uint4* s = (const uint4*)src;
  uint4* d = (uint4*)dst;
  int chunks = nelem >> 3;  // 16B per chunk
#if USE_ASYNC_LDS
  for (int i = t; i < chunks; i += 256)
    __builtin_amdgcn_global_load_async_to_lds_b128(
        (as1_v4i_ptr)(s + i), (as3_v4i_ptr)(d + i), 0, 0);
  wait_asynccnt0();
#else
  for (int i = t; i < chunks; i += 256) d[i] = s[i];
#endif
}

// A fragment (16x32 bf16 MxK) from a bf16 LDS row: two ds_load_b128 --------
// lane<16: row=lane,    K = k0+0..7  and k0+16..23
// lane>=16: row=lane-16, K = k0+8..15 and k0+24..31
struct v8us_pair { v8us lo, hi; };
__device__ __forceinline__ v16bf load_a_frag(const bf16_t* xrow, int k0, int lane) {
  int kb = k0 + ((lane & 16) ? 8 : 0);
  v8us_pair p;
  p.lo = *(const v8us*)(xrow + kb);
  p.hi = *(const v8us*)(xrow + kb + 16);
  return __builtin_bit_cast(v16bf, p);
}

// B fragment: pre-packed fragment-ordered bf16, 32B contiguous per lane ----
__device__ __forceinline__ v16bf load_b_frag(const bf16_t* wfrag,
                                             int kt, int nt, int lane) {
  const bf16_t* p = wfrag + (((size_t)(kt * 8 + nt) * 32 + lane) * 16);
  return *(const v16bf*)p;
}

__device__ __forceinline__ v8f wmma_bf16(v16bf a, v16bf b, v8f c) {
  return __builtin_amdgcn_wmma_f32_16x16x32_bf16(false, a, false, b,
                                                 (short)0, c, false, false);
}

// ---------------------------------------------------------------------------
// Weight packer: fp32 [Krows,128] -> fragment-ordered bf16, K zero-padded
// grid = (Kpad/32)*8 blocks x 32 threads
// ---------------------------------------------------------------------------
__global__ void pack_w_kernel(const float* __restrict__ W, int Krows,
                              bf16_t* __restrict__ dst) {
  int lane = threadIdx.x;
  int nt = blockIdx.x & 7;
  int kt = blockIdx.x >> 3;
  int col = nt * 16 + (lane & 15);
  int kb  = kt * 32 + ((lane & 16) ? 8 : 0);
  bf16_t* p = dst + (((size_t)(kt * 8 + nt) * 32 + lane) * 16);
#pragma unroll
  for (int j = 0; j < 16; ++j) {
    int k = kb + (j < 8 ? j : j + 8);
    float v = (k < Krows) ? W[(size_t)k * 128 + col] : 0.0f;
    p[j] = f2bf(v);
  }
}

__global__ void zero_kernel(float* __restrict__ p, int n) {
  for (int i = blockIdx.x * 256 + threadIdx.x; i < n; i += gridDim.x * 256)
    p[i] = 0.0f;
}

// ---------------------------------------------------------------------------
// Fingerprint update: softmax(X @ W + b) segment-summed into fp.
// Block = 256 threads (8 waves), 16 atoms x 128 features.
// mol_ids sorted -> reduce equal-mol runs in LDS, leaders do the atomics.
// ---------------------------------------------------------------------------
template <int KIN, typename XT>
__global__ __launch_bounds__(256) void fp_update_kernel(
    const XT* __restrict__ X, const bf16_t* __restrict__ Wfrag,
    const float* __restrict__ bias, const int* __restrict__ mol_ids,
    float* __restrict__ fp) {
  __shared__ __align__(16) bf16_t Xs[16 * KIN];
  __shared__ float Ls[16 * 128];
  __shared__ float Red[256];
  __shared__ int MolS[16];

  int t = threadIdx.x;
  int lane = t & 31, wave = t >> 5;
  int base = blockIdx.x * 16;

  if (t < 16) MolS[t] = mol_ids[base + t];
  stage_block(Xs, X + (size_t)base * KIN, 16 * KIN, t);
  __syncthreads();

  v8f acc = {};
  const bf16_t* xrow = &Xs[(lane & 15) * KIN];
#pragma unroll
  for (int kt = 0; kt < KIN / 32; ++kt) {
    v16bf a = load_a_frag(xrow, kt * 32, lane);
    v16bf b = load_b_frag(Wfrag, kt, wave, lane);
    acc = wmma_bf16(a, b, acc);
  }

  int col = wave * 16 + (lane & 15);
  int row0 = (lane & 16) ? 8 : 0;
  float bv = bias[col];
#pragma unroll
  for (int r = 0; r < 8; ++r) Ls[(row0 + r) * 128 + col] = acc[r] + bv;
  __syncthreads();

  // softmax per atom row: 16 threads per row, 8 features each
  int row = t >> 4, seg = t & 15;
  const float* lrow = &Ls[row * 128];
  float lv[8];
  float m = -3.0e38f;
#pragma unroll
  for (int j = 0; j < 8; ++j) { lv[j] = lrow[seg * 8 + j]; m = fmaxf(m, lv[j]); }
  Red[t] = m;
  __syncthreads();
  float rm = Red[row * 16];
#pragma unroll
  for (int j = 1; j < 16; ++j) rm = fmaxf(rm, Red[row * 16 + j]);
  __syncthreads();
  float s = 0.0f;
#pragma unroll
  for (int j = 0; j < 8; ++j) { lv[j] = __expf(lv[j] - rm); s += lv[j]; }
  Red[t] = s;
  __syncthreads();
  float rs = Red[row * 16];
#pragma unroll
  for (int j = 1; j < 16; ++j) rs += Red[row * 16 + j];
  float inv = 1.0f / rs;
#pragma unroll
  for (int j = 0; j < 8; ++j) lv[j] *= inv;

  // write normalized probs back, then run-leaders reduce + atomicAdd
  __syncthreads();
#pragma unroll
  for (int j = 0; j < 8; ++j) Ls[row * 128 + seg * 8 + j] = lv[j];
  __syncthreads();

  int molr = MolS[row];
  bool leader = (row == 0) || (MolS[row - 1] != molr);
  if (leader) {
    float a8[8];
#pragma unroll
    for (int j = 0; j < 8; ++j) a8[j] = lv[j];
    for (int rr = row + 1; rr < 16 && MolS[rr] == molr; ++rr)
#pragma unroll
      for (int j = 0; j < 8; ++j) a8[j] += Ls[rr * 128 + seg * 8 + j];
    float* dst = fp + (size_t)molr * 128 + seg * 8;
#pragma unroll
    for (int j = 0; j < 8; ++j) atomicAdd(dst + j, a8[j]);
  }
}

// ---------------------------------------------------------------------------
// Degree conv: x = X@selfW + concat(sum_nbr(X), sum_nbr(bond))@dW + bias
// Writes pre-BN x (fp32) to Xout, accumulates per-feature sum/sumsq.
// One launch per degree bucket; grid = ND/16 blocks of 256 threads.
// ---------------------------------------------------------------------------
template <int KIN, typename XT>
__global__ __launch_bounds__(256) void conv_kernel(
    const XT* __restrict__ X, const float* __restrict__ bond,
    const int* __restrict__ a_nbr, const int* __restrict__ b_nbr,
    const bf16_t* __restrict__ selfWfrag, const bf16_t* __restrict__ dWfrag,
    const float* __restrict__ bias, int deg, int bucket_base,
    float* __restrict__ Xout, float* __restrict__ stat_sum,
    float* __restrict__ stat_sq) {
  constexpr int KPAD = KIN + 32;  // 96 or 160 (K=KIN+16 zero-padded to x32)
  __shared__ __align__(16) bf16_t Ss[16 * KIN];
  __shared__ __align__(16) bf16_t Cs[16 * KPAD];
  __shared__ float Ls[16 * 128];

  int t = threadIdx.x;
  int lane = t & 31, wave = t >> 5;
  int i0 = blockIdx.x * 16;  // group offset inside bucket

  // self rows (contiguous)
  stage_block(Ss, X + (size_t)(bucket_base + i0) * KIN, 16 * KIN, t);
  // gathered neighbor-atom sums
  for (int idx = t; idx < 16 * KIN; idx += 256) {
    int r = idx / KIN, f = idx - r * KIN;
    const int* nb = a_nbr + (size_t)(i0 + r) * deg;
    float s = 0.0f;
    for (int j = 0; j < deg; ++j) s += xval(X, (size_t)nb[j] * KIN + f);
    Cs[r * KPAD + f] = f2bf(s);
  }
  // gathered bond sums + zero pad
  for (int idx = t; idx < 16 * 16; idx += 256) {
    int r = idx >> 4, f = idx & 15;
    const int* nb = b_nbr + (size_t)(i0 + r) * deg;
    float s = 0.0f;
    for (int j = 0; j < deg; ++j) s += bond[(size_t)nb[j] * EDGE_F + f];
    Cs[r * KPAD + KIN + f] = f2bf(s);
    Cs[r * KPAD + KIN + 16 + f] = 0;
  }
  __syncthreads();

  v8f acc = {};
  const bf16_t* srow = &Ss[(lane & 15) * KIN];
  const bf16_t* crow = &Cs[(lane & 15) * KPAD];
#pragma unroll
  for (int kt = 0; kt < KIN / 32; ++kt) {  // self path
    v16bf a = load_a_frag(srow, kt * 32, lane);
    v16bf b = load_b_frag(selfWfrag, kt, wave, lane);
    acc = wmma_bf16(a, b, acc);
  }
#pragma unroll
  for (int kt = 0; kt < KPAD / 32; ++kt) {  // neighbor path
    v16bf a = load_a_frag(crow, kt * 32, lane);
    v16bf b = load_b_frag(dWfrag, kt, wave, lane);
    acc = wmma_bf16(a, b, acc);
  }

  int col = wave * 16 + (lane & 15);
  int row0 = (lane & 16) ? 8 : 0;
  float bv = bias[col];
#pragma unroll
  for (int r = 0; r < 8; ++r) Ls[(row0 + r) * 128 + col] = acc[r] + bv;
  __syncthreads();

  // BatchNorm partial stats: one feature per thread (t < 128)
  if (t < 128) {
    float s = 0.0f, q = 0.0f;
#pragma unroll
    for (int r = 0; r < 16; ++r) {
      float v = Ls[r * 128 + t];
      s += v; q += v * v;
    }
    atomicAdd(&stat_sum[t], s);
    atomicAdd(&stat_sq[t], q);
  }
  // coalesced store of pre-BN x
  for (int idx = t; idx < 16 * 128; idx += 256) {
    int r = idx >> 7, f = idx & 127;
    Xout[(size_t)(bucket_base + i0 + r) * 128 + f] = Ls[idx];
  }
}

__global__ void bn_finalize_kernel(float* __restrict__ sum,
                                   float* __restrict__ sq, float n) {
  int f = threadIdx.x;
  float mean = sum[f] / n;
  float var = sq[f] / n - mean * mean;  // biased variance (jnp.var default)
  sum[f] = mean;
  sq[f] = rsqrtf(var + 1e-5f);
}

__global__ void bn_relu_kernel(const float* __restrict__ Xin,
                               const float* __restrict__ mean,
                               const float* __restrict__ inv,
                               bf16_t* __restrict__ Hout, int total) {
  for (int i = blockIdx.x * 256 + threadIdx.x; i < total; i += gridDim.x * 256) {
    int f = i & 127;
    float v = (Xin[i] - mean[f]) * inv[f];
    Hout[i] = f2bf(fmaxf(v, 0.0f));
  }
}

// ---------------------------------------------------------------------------
extern "C" void kernel_launch(void* const* d_in, const int* in_sizes, int n_in,
                              void* d_out, int out_size, void* d_ws,
                              size_t ws_size, hipStream_t stream) {
  (void)in_sizes; (void)n_in; (void)out_size; (void)ws_size;

  const float* atom = (const float*)d_in[0];
  const float* bond = (const float*)d_in[1];
  const int* a_nbr[4] = {(const int*)d_in[2], (const int*)d_in[4],
                         (const int*)d_in[6], (const int*)d_in[8]};
  const int* b_nbr[4] = {(const int*)d_in[3], (const int*)d_in[5],
                         (const int*)d_in[7], (const int*)d_in[9]};
  const int* mol_ids = (const int*)d_in[10];
  const float* out_W[3] = {(const float*)d_in[11], (const float*)d_in[13],
                           (const float*)d_in[15]};
  const float* out_b[3] = {(const float*)d_in[12], (const float*)d_in[14],
                           (const float*)d_in[16]};
  const float* c1_selfW = (const float*)d_in[17];
  const float* c1_bias  = (const float*)d_in[18];
  const float* c1_dW[4] = {(const float*)d_in[19], (const float*)d_in[20],
                           (const float*)d_in[21], (const float*)d_in[22]};
  const float* c2_selfW = (const float*)d_in[23];
  const float* c2_bias  = (const float*)d_in[24];
  const float* c2_dW[4] = {(const float*)d_in[25], (const float*)d_in[26],
                           (const float*)d_in[27], (const float*)d_in[28]};

  // workspace layout
  char* ws = (char*)d_ws;
  float*  Xscr = (float*)(ws);                    // [N,128] fp32 pre-BN (102.4MB)
  bf16_t* Hbuf = (bf16_t*)(ws + 102400000ull);    // [N,128] bf16 h1 then h2
  float*  sum1 = (float*)(ws + 204800000ull);     // 4 x 128 floats stats
  float*  sq1  = sum1 + 128;
  float*  sum2 = sum1 + 256;
  float*  sq2  = sum1 + 384;
  bf16_t* wp = (bf16_t*)(ws + 204802048ull);      // packed weights (384KB)
  auto seg = [&](int ktile_off) { return wp + (size_t)ktile_off * 4096; };
  bf16_t* W0p = seg(0);   // 2 ktiles (K=64)
  bf16_t* W1p = seg(2);   // 4 ktiles (K=128)
  bf16_t* W2p = seg(6);   // 4
  bf16_t* S1p = seg(10);  // 2
  bf16_t* S2p = seg(12);  // 4
  bf16_t* D1p[4] = {seg(16), seg(19), seg(22), seg(25)};  // 3 each (80->96)
  bf16_t* D2p[4] = {seg(28), seg(33), seg(38), seg(43)};  // 5 each (144->160)

  float* fp = (float*)d_out;

  // deterministic init every call (graph replay safe)
  zero_kernel<<<2048, 256, 0, stream>>>(fp, B_MOL * 128);
  zero_kernel<<<2, 256, 0, stream>>>(sum1, 512);

  // pack weights -> fragment-ordered bf16
  pack_w_kernel<<<dim3(2 * 8), 32, 0, stream>>>(out_W[0], 64, W0p);
  pack_w_kernel<<<dim3(4 * 8), 32, 0, stream>>>(out_W[1], 128, W1p);
  pack_w_kernel<<<dim3(4 * 8), 32, 0, stream>>>(out_W[2], 128, W2p);
  pack_w_kernel<<<dim3(2 * 8), 32, 0, stream>>>(c1_selfW, 64, S1p);
  pack_w_kernel<<<dim3(4 * 8), 32, 0, stream>>>(c2_selfW, 128, S2p);
  for (int d = 0; d < 4; ++d) {
    pack_w_kernel<<<dim3(3 * 8), 32, 0, stream>>>(c1_dW[d], 80, D1p[d]);
    pack_w_kernel<<<dim3(5 * 8), 32, 0, stream>>>(c2_dW[d], 144, D2p[d]);
  }

  // fp += segsum(softmax(atom @ W0 + b0))
  fp_update_kernel<64, float><<<N_ATOMS / 16, 256, 0, stream>>>(
      atom, W0p, out_b[0], mol_ids, fp);
  // conv layer 1 -> pre-BN x + stats
  for (int d = 0; d < 4; ++d)
    conv_kernel<64, float><<<ND_B / 16, 256, 0, stream>>>(
        atom, bond, a_nbr[d], b_nbr[d], S1p, D1p[d], c1_bias, d + 1,
        d * ND_B, Xscr, sum1, sq1);
  bn_finalize_kernel<<<1, 128, 0, stream>>>(sum1, sq1, (float)N_ATOMS);
  bn_relu_kernel<<<4096, 256, 0, stream>>>(Xscr, sum1, sq1, Hbuf,
                                           N_ATOMS * 128);
  // fp += segsum(softmax(h1 @ W1 + b1))
  fp_update_kernel<128, bf16_t><<<N_ATOMS / 16, 256, 0, stream>>>(
      Hbuf, W1p, out_b[1], mol_ids, fp);
  // conv layer 2
  for (int d = 0; d < 4; ++d)
    conv_kernel<128, bf16_t><<<ND_B / 16, 256, 0, stream>>>(
        Hbuf, bond, a_nbr[d], b_nbr[d], S2p, D2p[d], c2_bias, d + 1,
        d * ND_B, Xscr, sum2, sq2);
  bn_finalize_kernel<<<1, 128, 0, stream>>>(sum2, sq2, (float)N_ATOMS);
  bn_relu_kernel<<<4096, 256, 0, stream>>>(Xscr, sum2, sq2, Hbuf,
                                           N_ATOMS * 128);
  // fp += segsum(softmax(h2 @ W2 + b2))
  fp_update_kernel<128, bf16_t><<<N_ATOMS / 16, 256, 0, stream>>>(
      Hbuf, W2p, out_b[2], mol_ids, fp);
}
